// ConvAttention_11587821765255
// MI455X (gfx1250) — matrix operand order repaired and checked by reading, not compiled, not run on hardware
//
#include <hip/hip_runtime.h>

// ---------------- problem constants ----------------
#define BATCH 4
#define CDIM  64
#define DDIM  64
#define HDIM  256
#define WDIM  256
#define TH    16          // tile height (pixels)
#define TW    16          // tile width
#define RH    18          // halo region height
#define RW    18          // halo region width
#define RNUM  (RH * RW)   // 324 real region rows
#define RPAD  336         // padded to 21 * 16 for WMMA N-tiles (pixels)
#define MT    (RPAD / 16) // 21 pixel tiles
#define XST   72          // LDS row stride (bf16): 144B, 16B-aligned rows for b128 ops
#define WST   72          // transposed-weight LDS row stride (bf16)

typedef __bf16 bf16_t;
typedef __attribute__((ext_vector_type(16))) __bf16 v16bf;
typedef __attribute__((ext_vector_type(8)))  __bf16 v8bf;
typedef __attribute__((ext_vector_type(8)))  float  v8f;

union Frag {           // one WMMA 16-bf16 operand = two 16B LDS vector loads
    v16bf v;
    uint4 q[2];
};

__global__ __launch_bounds__(256) void conv_attn(
    const float* __restrict__ x,
    const float* __restrict__ Wq, const float* __restrict__ bq,
    const float* __restrict__ Wk, const float* __restrict__ bk,
    const float* __restrict__ Wv, const float* __restrict__ bv,
    float* __restrict__ out)
{
    // LDS: x halo + q/k/v (bf16, [pixel][d]) + transposed weights (bf16, [d][c]) + biases
    __shared__ alignas(16) bf16_t xs[RPAD * XST];
    __shared__ alignas(16) bf16_t qs[RPAD * XST];
    __shared__ alignas(16) bf16_t ks[RPAD * XST];
    __shared__ alignas(16) bf16_t vs[RPAD * XST];
    __shared__ alignas(16) bf16_t wt[3 * 64 * WST];
    __shared__ float  bsm[3 * 64];

    const int b   = blockIdx.z;
    const int h0  = blockIdx.y * TH;
    const int w0  = blockIdx.x * TW;
    const int tid = threadIdx.x;

    // prefetch the (heavily reused) weight matrices
    __builtin_prefetch(Wq + tid, 0, 0);
    __builtin_prefetch(Wk + tid, 0, 0);
    __builtin_prefetch(Wv + tid, 0, 0);

    // ---------------- phase 1: load x halo, transpose [C,H,W] -> [pixel][c], f32 -> bf16.
    // Each iteration handles one pixel row x 8 channels: 8 scalar global loads
    // (coalesced across lanes within a channel plane) -> one ds_store_b128.
    for (int e = tid; e < RPAD * (CDIM / 8); e += 256) {
        int cg = e / RPAD;          // channel group (8 channels)
        int r  = e % RPAD;          // consecutive lanes -> consecutive pixels
        union { uint4 u; bf16_t h[8]; } pk;
        if (r < RNUM) {
            int rh = r / RW, rw = r % RW;
            int gy = h0 - 1 + rh;
            int gx = w0 - 1 + rw;
            bool valid = (gy >= 0) && (gy < HDIM) && (gx >= 0) && (gx < WDIM);
            const float* xp = x + (((size_t)b * CDIM + cg * 8) * HDIM + gy) * WDIM + gx;
#pragma unroll
            for (int u = 0; u < 8; ++u)
                pk.h[u] = (bf16_t)(valid ? xp[(size_t)u * (HDIM * WDIM)] : 0.0f);
        } else {
            pk.u = make_uint4(0, 0, 0, 0);
        }
        *(uint4*)&xs[r * XST + cg * 8] = pk.u;
    }

    // ---------------- phase 2: stage weights TRANSPOSED (wt[d][c] = W[c][d]); this is the
    // WMMA A matrix (W^T) with contiguous 16B K-runs. Global reads stay coalesced.
    for (int e = tid; e < 3 * 64 * 64; e += 256) {
        int m   = e >> 12;
        int idx = e & 4095;                       // coalesced read of W[c][d]
        const float* Wp = (m == 0) ? Wq : (m == 1) ? Wk : Wv;
        int cc = idx >> 6, d = idx & 63;
        wt[(m * 64 + d) * WST + cc] = (bf16_t)Wp[idx];
    }
    if (tid < 192) {
        const float* bp = (tid < 64) ? bq : (tid < 128) ? bk : bv;
        bsm[tid] = bp[tid & 63];
    }
    __syncthreads();

    // ---------------- phase 3: QKV projections as D = W^T (A, 16x32) x x^T (B, 32x16)
    // with v_wmma_f32_16x16x32_bf16. D is [d rows, pixel cols]: each lane's 8
    // accumulators are 8 consecutive d of ONE pixel -> one packed ds_store_b128.
    const int wave  = tid >> 5;
    const int lane  = tid & 31;
    const int lrow  = lane & 15;   // A: M row (d); B/D: N col (pixel)
    const int lhalf = lane >> 4;

    for (int task = wave; task < MT * 3; task += 8) {
        int mt  = task / 3;                // pixel tile
        int mat = task % 3;
        int prow = mt * 16 + lrow;         // this lane's pixel (B column / D column)

        // B fragment (32x16): lane = pixel column; K = 16*half..+15 contiguous in xs row.
        const bf16_t* brow = &xs[prow * XST];
        Frag b0, b1;
        b0.q[0] = *(const uint4*)(brow + lhalf * 16);
        b0.q[1] = *(const uint4*)(brow + lhalf * 16 + 8);
        b1.q[0] = *(const uint4*)(brow + 32 + lhalf * 16);
        b1.q[1] = *(const uint4*)(brow + 32 + lhalf * 16 + 8);

        bf16_t*      outb = (mat == 0) ? qs : (mat == 1) ? ks : vs;
        const float* bb   = &bsm[mat * 64];

        for (int dt = 0; dt < 4; ++dt) {
            // A fragment (16x32): lane row = d; K = 16*half..+15 contiguous in wt row.
            const bf16_t* arow = &wt[(mat * 64 + dt * 16 + lrow) * WST];
            Frag a0, a1;
            a0.q[0] = *(const uint4*)(arow + lhalf * 16);
            a0.q[1] = *(const uint4*)(arow + lhalf * 16 + 8);
            a1.q[0] = *(const uint4*)(arow + 32 + lhalf * 16);
            a1.q[1] = *(const uint4*)(arow + 32 + lhalf * 16 + 8);

            // bias depends on M (= d): c[i] = b[dt*16 + 8*half + i]
            v8f c;
#pragma unroll
            for (int i = 0; i < 8; ++i) c[i] = bb[dt * 16 + lhalf * 8 + i];

            c = __builtin_amdgcn_wmma_f32_16x16x32_bf16(false, a0.v, false, b0.v,
                                                        (short)0, c, false, false);
            c = __builtin_amdgcn_wmma_f32_16x16x32_bf16(false, a1.v, false, b1.v,
                                                        (short)0, c, false, false);

            // D: lane's 8 values = d = dt*16 + 8*half + 0..7 of pixel `prow`:
            // pack to 8 bf16 and store as one b128 (offset is 16B aligned).
            union { uint4 u; bf16_t h[8]; } pk;
#pragma unroll
            for (int i = 0; i < 8; ++i) pk.h[i] = (bf16_t)c[i];
            *(uint4*)&outb[prow * XST + dt * 16 + lhalf * 8] = pk.u;
        }
    }
    __syncthreads();

    // ---------------- phase 4: 3x3 neighborhood attention, one thread per pixel.
    // All q/k/v reads are 16B ds_load_b128 of contiguous bf16 rows.
    {
        int ph = tid >> 4, pw = tid & 15;
        int gy = h0 + ph, gx = w0 + pw;
        int rc = (ph + 1) * RW + (pw + 1);

        float qv[64];
        const v8bf* qrow = (const v8bf*)&qs[rc * XST];
#pragma unroll
        for (int j = 0; j < 8; ++j) {
            v8bf t = qrow[j];
#pragma unroll
            for (int i = 0; i < 8; ++i) qv[8 * j + i] = (float)t[i];
        }

        float sc[9];
        float smax = -1e30f;
#pragma unroll
        for (int n = 0; n < 9; ++n) {
            int dy = n / 3 - 1, dx = n % 3 - 1;
            bool valid = (gy + dy >= 0) && (gy + dy < HDIM) &&
                         (gx + dx >= 0) && (gx + dx < WDIM);
            int rn = rc + dy * RW + dx;
            const v8bf* krow = (const v8bf*)&ks[rn * XST];
            float s = 0.0f;
#pragma unroll
            for (int j = 0; j < 8; ++j) {
                v8bf t = krow[j];
#pragma unroll
                for (int i = 0; i < 8; ++i) s += qv[8 * j + i] * (float)t[i];
            }
            s = valid ? s * 0.125f : -1e9f;   // 1/sqrt(64), masked like reference
            sc[n] = s;
            smax = fmaxf(smax, s);
        }
        float ssum = 0.0f;
#pragma unroll
        for (int n = 0; n < 9; ++n) { sc[n] = __expf(sc[n] - smax); ssum += sc[n]; }
        float inv = 1.0f / ssum;              // center is always valid -> ssum >= 1
#pragma unroll
        for (int n = 0; n < 9; ++n) sc[n] *= inv;

        // out[b, d, gy, gx]; consecutive lanes = consecutive gx -> coalesced per-d
        size_t obase = (((size_t)b * DDIM) * HDIM + gy) * WDIM + gx;
#pragma unroll
        for (int j = 0; j < 8; ++j) {
            float acc[8];
#pragma unroll
            for (int i = 0; i < 8; ++i) acc[i] = 0.0f;
#pragma unroll
            for (int n = 0; n < 9; ++n) {
                int rn = rc + (n / 3 - 1) * RW + (n % 3 - 1);
                v8bf t = ((const v8bf*)&vs[rn * XST])[j];
#pragma unroll
                for (int i = 0; i < 8; ++i) acc[i] += sc[n] * (float)t[i];
            }
#pragma unroll
            for (int i = 0; i < 8; ++i)
                out[obase + (size_t)(8 * j + i) * (HDIM * WDIM)] = acc[i];
        }
    }
}

extern "C" void kernel_launch(void* const* d_in, const int* in_sizes, int n_in,
                              void* d_out, int out_size, void* d_ws, size_t ws_size,
                              hipStream_t stream) {
    const float* x  = (const float*)d_in[0];
    const float* Wq = (const float*)d_in[1];
    const float* bq = (const float*)d_in[2];
    const float* Wk = (const float*)d_in[3];
    const float* bk = (const float*)d_in[4];
    const float* Wv = (const float*)d_in[5];
    const float* bv = (const float*)d_in[6];
    float* out = (float*)d_out;

    dim3 grid(WDIM / TW, HDIM / TH, BATCH);
    conv_attn<<<grid, dim3(256, 1, 1), 0, stream>>>(x, Wq, bq, Wk, bk, Wv, bv, out);
}
